// LocalGatedKL_45518063403694
// MI455X (gfx1250) — matrix operand order repaired and checked by reading, not compiled, not run on hardware
//
#include <hip/hip_runtime.h>
#include <hip/hip_bf16.h>

// ---------------------------------------------------------------------------
// LocalGatedKL on gfx1250 (MI455X)
//   fs3 [8,256,64,64] f32, ft [8,512,32,32] f32, Ws [64,256], Wt [64,512]
//   out: scalar f32
// Pipeline:
//   0) pack Ws/Wt into bf16 WMMA A-fragment order (one-time, tiny)
//   1) t_small = Wt @ ft            (WMMA bf16 GEMM, 64x512x8192)
//   2) s       = Ws @ fs3           (WMMA bf16 GEMM, 64x256x32768)
//   3) t = bilinear2x(t_small)      (exact: upsample commutes with 1x1 conv)
//   4) per-pixel L2 normalize t (also channel-mean for Sobel), normalize s
//   5) Sobel magnitude + per-image min/max (int-bit atomics, exact)
//   6) 24-neighbor affinities, log-softmax(tau=0.1), entropy & edge gating,
//      masked KL, block partial sums
//   7) deterministic final reduce -> out[0]
// GEMM mapping: one wave = one 16-pixel tile, B fragment loaded once and
// reused against all 4 weight-row tiles (4 accumulators, 4 WMMA / k-step).
// ---------------------------------------------------------------------------

typedef __attribute__((ext_vector_type(16))) __bf16          v16bf;
typedef __attribute__((ext_vector_type(16))) unsigned short  v16u;
typedef __attribute__((ext_vector_type(8)))  float           v8f;

#define BATCH   8
#define DCH     64      // projected channels
#define PIX     4096    // 64*64
#define PIX_SM  1024    // 32*32
#define TAU_INV 10.0f
#define MAXH_F  3.17805383035f  // log(24)

__device__ __forceinline__ unsigned short f32_to_bf16_rne(float x) {
  unsigned int u = __float_as_uint(x);
  u += 0x7FFFu + ((u >> 16) & 1u);   // round-to-nearest-even truncation
  return (unsigned short)(u >> 16);
}

// ---------------------------------------------------------------------------
// Pack W [64,K] f32 -> bf16 A-fragments, layout [(mTile*K/32 + kstep)*32 + lane][16]
// per CDNA5 ISA 7.12.2 16-bit A 16x32 layout.
// ---------------------------------------------------------------------------
__global__ void pack_w_kernel(const float* __restrict__ W,
                              unsigned short* __restrict__ Wpk, int K)
{
  const int tid = blockIdx.x * blockDim.x + threadIdx.x;
  const int nSteps = K >> 5;
  if (tid >= 4 * nSteps * 32) return;
  const int  lane  = tid & 31;
  const int  kstep = (tid >> 5) % nSteps;
  const int  mTile = (tid >> 5) / nSteps;
  const int  col   = lane & 15;
  const bool hi    = lane >= 16;
  const float* Wrow = W + (size_t)(mTile * 16 + col) * K;
  unsigned short* o = Wpk + (size_t)tid * 16;
#pragma unroll
  for (int e = 0; e < 16; ++e) {
    const int vg = e >> 1, ps = e & 1;
    const int koff = (vg < 4) ? ((hi ? 8 : 0) + (vg << 1) + ps)
                              : ((hi ? 24 : 16) + ((vg - 4) << 1) + ps);
    o[e] = f32_to_bf16_rne(Wrow[(kstep << 5) + koff]);
  }
}

// ---------------------------------------------------------------------------
// WMMA projection GEMM:  Y[b, m, p] = sum_k W[m,k] * X[b,k,p],  M = 64
// One wave per 16-pixel tile; B fragment shared across 4 mTiles.
// ---------------------------------------------------------------------------
__global__ __launch_bounds__(128) void proj_wmma_kernel(
    const unsigned short* __restrict__ Wpk,  // packed A fragments
    const float* __restrict__ X,             // [BATCH, K, P]
    float* __restrict__ Y,                   // [BATCH, 64, P]
    int K, int P)
{
  const int  lane  = threadIdx.x & 31;
  const int  wv    = threadIdx.x >> 5;
  const int  nTile = blockIdx.x * 4 + wv;     // pixel tile (global)
  const int  col   = lane & 15;
  const bool hi    = lane >= 16;

  const int gp  = nTile << 4;
  const int b   = gp / P;
  const int pix = gp - b * P + col;
  const int nSteps = K >> 5;

  const float* Xb  = X + (size_t)b * K * P;
  const v16u*  Apk = (const v16u*)Wpk;        // 32B vector per (mTile,kstep,lane)

  v8f acc[4];
#pragma unroll
  for (int mt = 0; mt < 4; ++mt)
    acc[mt] = (v8f){0.f, 0.f, 0.f, 0.f, 0.f, 0.f, 0.f, 0.f};

#pragma unroll 2
  for (int ks = 0; ks < nSteps; ++ks) {
    const float* Xcol = Xb + (size_t)((ks << 5) + (hi ? 8 : 0)) * P + pix;
    if (ks + 1 < nSteps)
      __builtin_prefetch(Xcol + (size_t)32 * P, 0, 0);  // global_prefetch_b8

    // B fragment: col = pixel, K table order (lanes<16: k0..7,16..23; else k8..15,24..31)
    v16u bu;
#pragma unroll
    for (int e = 0; e < 16; ++e)
      bu[e] = f32_to_bf16_rne(Xcol[(size_t)(e + (e & 8)) * P]);
    const v16bf bm = __builtin_bit_cast(v16bf, bu);

#pragma unroll
    for (int mt = 0; mt < 4; ++mt) {
      const v16bf a = __builtin_bit_cast(
          v16bf, Apk[(size_t)(mt * nSteps + ks) * 32 + lane]);
      acc[mt] = __builtin_amdgcn_wmma_f32_16x16x32_bf16(
          false, a, false, bm, (short)0, acc[mt], false, false);
    }
  }

  // D layout: lane<16 -> rows mBase+0..7; lane>=16 -> rows mBase+8..15
#pragma unroll
  for (int mt = 0; mt < 4; ++mt) {
    float* Yp = Y + (size_t)b * DCH * P
                  + (size_t)(mt * 16 + (hi ? 8 : 0)) * P + pix;
#pragma unroll
    for (int r = 0; r < 8; ++r) Yp[(size_t)r * P] = acc[mt][r];
  }
}

// ---------------------------------------------------------------------------
// Bilinear 2x upsample, half-pixel centers (matches jax.image.resize bilinear)
// ---------------------------------------------------------------------------
__global__ void upsample2x_kernel(const float* __restrict__ src,  // [8,64,32,32]
                                  float* __restrict__ dst)        // [8,64,64,64]
{
  const int idx = blockIdx.x * blockDim.x + threadIdx.x;
  if (idx >= BATCH * DCH * PIX) return;
  const int w = idx & 63, h = (idx >> 6) & 63;
  const int plane = idx >> 12;                  // b*64 + c
  const float sy = h * 0.5f - 0.25f;
  const float sx = w * 0.5f - 0.25f;
  int y0 = (int)floorf(sy), x0 = (int)floorf(sx);
  const float fy = sy - y0, fx = sx - x0;
  const int y1 = min(y0 + 1, 31), x1 = min(x0 + 1, 31);
  y0 = max(y0, 0); x0 = max(x0, 0);
  const float* sp = src + (size_t)plane * PIX_SM;
  const float v00 = sp[y0 * 32 + x0], v01 = sp[y0 * 32 + x1];
  const float v10 = sp[y1 * 32 + x0], v11 = sp[y1 * 32 + x1];
  dst[idx] = (1.f - fy) * ((1.f - fx) * v00 + fx * v01)
           +        fy  * ((1.f - fx) * v10 + fx * v11);
}

// ---------------------------------------------------------------------------
// Per-pixel channel L2 normalize (in place); optional channel mean (raw data)
// ---------------------------------------------------------------------------
__global__ void normalize_kernel(float* __restrict__ f,          // [8,64,P]
                                 float* __restrict__ mean_out,   // [8,P] or null
                                 int P)
{
  const int idx = blockIdx.x * blockDim.x + threadIdx.x;
  if (idx >= BATCH * P) return;
  const int b = idx / P, p = idx - b * P;
  float* colp = f + (size_t)b * DCH * P + p;
  float ss = 0.f, sm = 0.f;
#pragma unroll 4
  for (int c = 0; c < DCH; ++c) {
    const float v = colp[(size_t)c * P];
    sm += v; ss += v * v;
  }
  if (mean_out) mean_out[idx] = sm * (1.0f / DCH);
  const float inv = 1.0f / fmaxf(sqrtf(ss), 1e-12f);
#pragma unroll 4
  for (int c = 0; c < DCH; ++c) colp[(size_t)c * P] *= inv;
}

// ---------------------------------------------------------------------------
// Sobel magnitude (zero-padded SAME) + per-image min/max via int-bit atomics
// ---------------------------------------------------------------------------
__global__ void sobel_kernel(const float* __restrict__ xm,   // [8,64,64]
                             float* __restrict__ mag,        // [8,4096]
                             int* __restrict__ minmax)       // [8][2]
{
  const int idx = blockIdx.x * blockDim.x + threadIdx.x;     // 32768
  const int b = idx >> 12, p = idx & (PIX - 1);
  const int h = p >> 6, w = p & 63;
  const float* X = xm + ((size_t)b << 12);
  auto at = [&](int y, int x) -> float {
    return (y < 0 || y > 63 || x < 0 || x > 63) ? 0.f : X[y * 64 + x];
  };
  const float gx = -at(h-1,w-1) + at(h-1,w+1) - 2.f*at(h,w-1) + 2.f*at(h,w+1)
                   -at(h+1,w-1) + at(h+1,w+1);
  const float gy = -at(h-1,w-1) - 2.f*at(h-1,w) - at(h-1,w+1)
                   +at(h+1,w-1) + 2.f*at(h+1,w) + at(h+1,w+1);
  const float m = sqrtf(gx * gx + gy * gy);
  mag[idx] = m;

  __shared__ float smin[256], smax[256];
  const int t = threadIdx.x;
  smin[t] = m; smax[t] = m;
  __syncthreads();
  for (int off = 128; off > 0; off >>= 1) {
    if (t < off) {
      smin[t] = fminf(smin[t], smin[t + off]);
      smax[t] = fmaxf(smax[t], smax[t + off]);
    }
    __syncthreads();
  }
  if (t == 0) {   // block lies entirely inside one image (4096 % 256 == 0)
    atomicMin(&minmax[2 * b],     __float_as_int(smin[0]));
    atomicMax(&minmax[2 * b + 1], __float_as_int(smax[0]));
  }
}

// ---------------------------------------------------------------------------
// Affinities + gated KL; per-block partial sums (deterministic)
// ---------------------------------------------------------------------------
__device__ __forceinline__ void affinities24(const float* __restrict__ Fb,
                                             int p, const int* __restrict__ q,
                                             float* __restrict__ aff)
{
#pragma unroll
  for (int j = 0; j < 24; ++j) aff[j] = 0.f;
  for (int c = 0; c < DCH; ++c) {
    const float* Fc = Fb + (size_t)c * PIX;
    const float fc = Fc[p];
#pragma unroll
    for (int j = 0; j < 24; ++j) aff[j] += fc * Fc[q[j]];
  }
}

__device__ __forceinline__ void log_softmax24(float* __restrict__ a)
{
  float mx = -1e30f;
#pragma unroll
  for (int j = 0; j < 24; ++j) { a[j] *= TAU_INV; mx = fmaxf(mx, a[j]); }
  float s = 0.f;
#pragma unroll
  for (int j = 0; j < 24; ++j) s += __expf(a[j] - mx);
  const float lse = mx + __logf(s);
#pragma unroll
  for (int j = 0; j < 24; ++j) a[j] -= lse;
}

__global__ void loss_kernel(const float* __restrict__ tn,    // normalized t
                            const float* __restrict__ sn,    // normalized s
                            const float* __restrict__ mag,   // raw sobel
                            const int*   __restrict__ minmax,
                            float* __restrict__ blockSums)   // [nblk*2]
{
  const int idx = blockIdx.x * blockDim.x + threadIdx.x;     // 32768
  const int b = idx >> 12, p = idx & (PIX - 1);
  const int h = p >> 6, w = p & 63;

  // reflect-padded neighbor offsets (center excluded)
  int q[24];
  {
    int ni = 0;
#pragma unroll
    for (int di = -2; di <= 2; ++di)
#pragma unroll
      for (int dj = -2; dj <= 2; ++dj) {
        if (di == 0 && dj == 0) continue;
        int y = h + di; y = y < 0 ? -y : (y > 63 ? 126 - y : y);
        int x = w + dj; x = x < 0 ? -x : (x > 63 ? 126 - x : x);
        q[ni++] = y * 64 + x;
      }
  }

  const float* Tb = tn + (size_t)b * DCH * PIX;
  const float* Sb = sn + (size_t)b * DCH * PIX;

  float lt[24], ls[24];
  affinities24(Tb, p, q, lt);
  affinities24(Sb, p, q, ls);
  log_softmax24(lt);
  log_softmax24(ls);

  float ent = 0.f, kl = 0.f;
#pragma unroll
  for (int j = 0; j < 24; ++j) {
    const float pj = __expf(lt[j]);
    ent -= pj * lt[j];
    kl  += pj * (lt[j] - ls[j]);
  }

  float wconf = 1.0f - ent / (MAXH_F + 1e-12f);
  wconf = fminf(fmaxf(wconf, 0.f), 1.f);
  const float mn  = __int_as_float(minmax[2 * b]);
  const float mxv = __int_as_float(minmax[2 * b + 1]);
  const float wedge = (mag[idx] - mn) / (mxv - mn + 1e-6f);
  float wgt = wconf * wconf * (1.0f + wedge);           // ALPHA=2, BETA=1
  const bool interior = (h >= 2) && (h < 62) && (w >= 2) && (w < 62);
  if (!interior) wgt = 0.f;

  // deterministic block reduction
  __shared__ float s0[256], s1[256];
  const int t = threadIdx.x;
  s0[t] = wgt * kl; s1[t] = wgt;
  __syncthreads();
  for (int off = 128; off > 0; off >>= 1) {
    if (t < off) { s0[t] += s0[t + off]; s1[t] += s1[t + off]; }
    __syncthreads();
  }
  if (t == 0) {
    blockSums[2 * blockIdx.x]     = s0[0];
    blockSums[2 * blockIdx.x + 1] = s1[0];
  }
}

__global__ void init_kernel(int* __restrict__ minmax)
{
  if (threadIdx.x < 2 * BATCH)
    minmax[threadIdx.x] = (threadIdx.x & 1) ? 0 : 0x7f7fffff;  // max<-0, min<-FLT_MAX
}

__global__ void finalize_kernel(const float* __restrict__ blockSums,
                                int nBlocks, float* __restrict__ out)
{
  __shared__ float s0[128], s1[128];
  const int t = threadIdx.x;
  float a = 0.f, c = 0.f;
  for (int i = t; i < nBlocks; i += 128) { a += blockSums[2*i]; c += blockSums[2*i+1]; }
  s0[t] = a; s1[t] = c;
  __syncthreads();
  for (int off = 64; off > 0; off >>= 1) {
    if (t < off) { s0[t] += s0[t + off]; s1[t] += s1[t + off]; }
    __syncthreads();
  }
  if (t == 0) out[0] = s0[0] / (s1[0] + 1e-6f);
}

// ---------------------------------------------------------------------------
extern "C" void kernel_launch(void* const* d_in, const int* in_sizes, int n_in,
                              void* d_out, int out_size, void* d_ws, size_t ws_size,
                              hipStream_t stream)
{
  const float* fs3 = (const float*)d_in[0];   // [8,256,64,64]
  const float* ft  = (const float*)d_in[1];   // [8,512,32,32]
  const float* Ws  = (const float*)d_in[2];   // [64,256]
  const float* Wt  = (const float*)d_in[3];   // [64,512]
  float* out = (float*)d_out;

  // workspace layout (floats)
  float* base    = (float*)d_ws;
  float* t_small = base;                                    //  524288 f
  float* s_buf   = t_small + (size_t)BATCH * DCH * PIX_SM;  // 2097152 f
  float* t_buf   = s_buf   + (size_t)BATCH * DCH * PIX;     // 2097152 f
  float* tmean   = t_buf   + (size_t)BATCH * DCH * PIX;     //   32768 f
  float* mag     = tmean   + (size_t)BATCH * PIX;           //   32768 f
  int*   minmax  = (int*)(mag + (size_t)BATCH * PIX);       //      16 i
  float* bsums   = (float*)(minmax + 2 * BATCH);            //     256 f
  unsigned short* wt_pk = (unsigned short*)(bsums + 256);   //   32768 u16
  unsigned short* ws_pk = wt_pk + 32768;                    //   16384 u16

  init_kernel<<<1, 32, 0, stream>>>(minmax);

  // pack weights into bf16 WMMA A-fragment order
  pack_w_kernel<<<8, 256, 0, stream>>>(Wt, wt_pk, 512);  // 4*16*32 = 2048 thr
  pack_w_kernel<<<4, 256, 0, stream>>>(Ws, ws_pk, 256);  // 4*8*32  = 1024 thr

  // t_small = Wt @ ft  : K=512, P=1024, 512 tiles / 4 waves = 128 blocks
  proj_wmma_kernel<<<128, 128, 0, stream>>>(wt_pk, ft, t_small, 512, PIX_SM);
  // s = Ws @ fs3       : K=256, P=4096, 2048 tiles / 4 waves = 512 blocks
  proj_wmma_kernel<<<512, 128, 0, stream>>>(ws_pk, fs3, s_buf, 256, PIX);

  // t = upsample2x(t_small)
  upsample2x_kernel<<<(BATCH * DCH * PIX + 255) / 256, 256, 0, stream>>>(t_small, t_buf);

  // normalize (t also emits channel mean of raw t for Sobel)
  normalize_kernel<<<(BATCH * PIX + 255) / 256, 256, 0, stream>>>(t_buf, tmean, PIX);
  normalize_kernel<<<(BATCH * PIX + 255) / 256, 256, 0, stream>>>(s_buf, nullptr, PIX);

  sobel_kernel<<<BATCH * PIX / 256, 256, 0, stream>>>(tmean, mag, minmax);

  const int nblk = BATCH * PIX / 256;   // 128
  loss_kernel<<<nblk, 256, 0, stream>>>(t_buf, s_buf, mag, minmax, bsums);

  finalize_kernel<<<1, 128, 0, stream>>>(bsums, nblk, out);
}
